// APPNP_84413287235626
// MI455X (gfx1250) — matrix operand order repaired and checked by reading, not compiled, not run on hardware
//
#include <hip/hip_runtime.h>

// ---------------- problem constants (match reference) ----------------
#define DIM_IN  771
#define KPAD1   800          // 771 padded to multiple of 32 (zero fill)
#define HID     256
#define N_GRAPH 512
#define KSTEPS  10
#define ALPHA   0.1f
#define BN_EPS  1e-5f

typedef __bf16 bf16;
typedef __attribute__((ext_vector_type(16))) __bf16 v16bf;
typedef __attribute__((ext_vector_type(8)))  __bf16 bf8v;
typedef __attribute__((ext_vector_type(8)))  float  v8f;
typedef __attribute__((ext_vector_type(4)))  float  f4;

// ---------------- utility ----------------
__global__ void k_zero_i32(int* __restrict__ p, int n) {
    for (int i = blockIdx.x * blockDim.x + threadIdx.x; i < n;
         i += gridDim.x * blockDim.x) p[i] = 0;
}

// x [N,771] f32 -> xb [N,800] bf16 zero-padded
__global__ void k_cvt_pad_x(const float* __restrict__ x, bf16* __restrict__ xb, int n) {
    for (int i = blockIdx.x * blockDim.x + threadIdx.x; i < n;
         i += gridDim.x * blockDim.x) {
        int r = i / KPAD1, k = i - r * KPAD1;
        xb[i] = (k < DIM_IN) ? (bf16)x[(size_t)r * DIM_IN + k] : (bf16)0.0f;
    }
}

// W [kin,nout] f32 -> out [nout,kpad] bf16 (transposed, zero-padded in K)
__global__ void k_packT(const float* __restrict__ W, bf16* __restrict__ out,
                        int kin, int nout, int kpad) {
    int total = nout * kpad;
    for (int i = blockIdx.x * blockDim.x + threadIdx.x; i < total;
         i += gridDim.x * blockDim.x) {
        int n = i / kpad, k = i - n * kpad;
        out[i] = (k < kin) ? (bf16)W[(size_t)k * nout + n] : (bf16)0.0f;
    }
}

// ---------------- WMMA GEMM: C[M,256] = A[M,lda] @ Bt^T (+bias) ----------------
// A bf16 row-major (lda = padded K), Bt bf16 [256, ldb] (row n = output column n).
// One wave computes a 16x64 slab: 4 x v_wmma_f32_16x16x32_bf16 per K-chunk.
__global__ void k_gemm_wmma(const bf16* __restrict__ A, int lda,
                            const bf16* __restrict__ Bt, int ldb,
                            const float* __restrict__ bias,
                            float* __restrict__ C, int M, int kchunks) {
    int wid  = (int)((blockIdx.x * blockDim.x + threadIdx.x) >> 5);
    int lane = threadIdx.x & 31;
    int mblk = wid >> 2, ngrp = wid & 3;
    int m0 = mblk * 16;
    if (m0 >= M) return;
    int lrow = lane & 15, lhi = lane >> 4;

    const bf16* arow = A + (size_t)(m0 + lrow) * lda + lhi * 8;
    const bf16* bptr[4];
#pragma unroll
    for (int t = 0; t < 4; ++t)
        bptr[t] = Bt + (size_t)(ngrp * 64 + t * 16 + lrow) * ldb + lhi * 8;

    v8f acc[4] = {};
    for (int kc = 0; kc < kchunks; ++kc) {
        int k0 = kc * 32;
        union { bf8v h[2]; v16bf v; } ua;
        ua.h[0] = *(const bf8v*)(arow + k0);        // K = khalf+0..7
        ua.h[1] = *(const bf8v*)(arow + k0 + 16);   // K = khalf+16..23
#pragma unroll
        for (int t = 0; t < 4; ++t) {
            union { bf8v h[2]; v16bf v; } ub;
            ub.h[0] = *(const bf8v*)(bptr[t] + k0);     // K = lhi*8+0..7
            ub.h[1] = *(const bf8v*)(bptr[t] + k0 + 16); // K = lhi*8+16..23
            acc[t] = __builtin_amdgcn_wmma_f32_16x16x32_bf16(
                false, ua.v, false, ub.v, (short)0, acc[t], false, false);
        }
    }
    int crow = m0 + lhi * 8;
#pragma unroll
    for (int t = 0; t < 4; ++t) {
        int col = ngrp * 64 + t * 16 + lrow;
        float bv = bias[col];
#pragma unroll
        for (int r = 0; r < 8; ++r)
            C[(size_t)(crow + r) * HID + col] = acc[t][r] + bv;
    }
}

// ---------------- BatchNorm stats: deterministic 2-stage column reduction ----------------
#define STAT_ROWS 256
__global__ void k_colstats_part(const float* __restrict__ C, int M,
                                float* __restrict__ ps, float* __restrict__ pq) {
    int f = threadIdx.x;
    int r0 = blockIdx.x * STAT_ROWS, r1 = r0 + STAT_ROWS;
    if (r1 > M) r1 = M;
    float s = 0.f, q = 0.f;
    for (int r = r0; r < r1; ++r) {
        float v = C[(size_t)r * HID + f];
        s += v; q += v * v;
    }
    ps[(size_t)blockIdx.x * HID + f] = s;
    pq[(size_t)blockIdx.x * HID + f] = q;
}

__global__ void k_colstats_final(const float* __restrict__ ps, const float* __restrict__ pq,
                                 int nb, int M, float* __restrict__ mu, float* __restrict__ rinv) {
    int f = blockIdx.x * blockDim.x + threadIdx.x;
    if (f >= HID) return;
    float s = 0.f, q = 0.f;
    for (int b = 0; b < nb; ++b) { s += ps[(size_t)b * HID + f]; q += pq[(size_t)b * HID + f]; }
    float m = s / (float)M;
    float var = q / (float)M - m * m;
    mu[f] = m;
    rinv[f] = rsqrtf(var + BN_EPS);
}

__global__ void k_bnrelu_bf16(const float* __restrict__ C, const float* __restrict__ mu,
                              const float* __restrict__ rinv, const float* __restrict__ g,
                              const float* __restrict__ bt, bf16* __restrict__ out, int n) {
    for (int i = blockIdx.x * blockDim.x + threadIdx.x; i < n;
         i += gridDim.x * blockDim.x) {
        int f = i & (HID - 1);
        float v = (C[i] - mu[f]) * rinv[f] * g[f] + bt[f];
        out[i] = (bf16)fmaxf(v, 0.0f);
    }
}

__global__ void k_bnrelu_f32(const float* __restrict__ C, const float* __restrict__ mu,
                             const float* __restrict__ rinv, const float* __restrict__ g,
                             const float* __restrict__ bt, float* __restrict__ out, int n) {
    for (int i = blockIdx.x * blockDim.x + threadIdx.x; i < n;
         i += gridDim.x * blockDim.x) {
        int f = i & (HID - 1);
        float v = (C[i] - mu[f]) * rinv[f] * g[f] + bt[f];
        out[i] = fmaxf(v, 0.0f);
    }
}

// ---------------- CSR build (dst-indexed) ----------------
__global__ void k_count_in(const int* __restrict__ dst, int e, int* __restrict__ cnt) {
    for (int i = blockIdx.x * blockDim.x + threadIdx.x; i < e;
         i += gridDim.x * blockDim.x) atomicAdd(&cnt[dst[i]], 1);
}

__global__ void k_dinv(const int* __restrict__ cnt, float* __restrict__ dinv, int n) {
    int i = blockIdx.x * blockDim.x + threadIdx.x;
    if (i < n) dinv[i] = rsqrtf((float)(cnt[i] + 1));   // +1 = self loop
}

__global__ void k_scanA(const int* __restrict__ cnt, int n,
                        int* __restrict__ excl, int* __restrict__ bsum) {
    __shared__ int sm[256];
    int i = blockIdx.x * 256 + threadIdx.x;
    int v = (i < n) ? cnt[i] : 0;
    sm[threadIdx.x] = v;
    __syncthreads();
    for (int off = 1; off < 256; off <<= 1) {
        int t = (threadIdx.x >= off) ? sm[threadIdx.x - off] : 0;
        __syncthreads();
        sm[threadIdx.x] += t;
        __syncthreads();
    }
    int incl = sm[threadIdx.x];
    if (i < n) excl[i] = incl - v;
    if (threadIdx.x == 255) bsum[blockIdx.x] = incl;
}

__global__ void k_scanB(int* __restrict__ bsum, int nb) {
    if (blockIdx.x == 0 && threadIdx.x == 0) {
        int run = 0;
        for (int b = 0; b < nb; ++b) { int t = bsum[b]; bsum[b] = run; run += t; }
    }
}

__global__ void k_scanC(int* __restrict__ excl, const int* __restrict__ bsum, int n) {
    int i = blockIdx.x * blockDim.x + threadIdx.x;
    if (i < n) excl[i] += bsum[i >> 8];
}

__global__ void k_fill(const int* __restrict__ src, const int* __restrict__ dst, int e,
                       const int* __restrict__ offs, int* __restrict__ cur,
                       const float* __restrict__ dinv,
                       int* __restrict__ csrc, float* __restrict__ cw) {
    for (int i = blockIdx.x * blockDim.x + threadIdx.x; i < e;
         i += gridDim.x * blockDim.x) {
        int s = src[i], d = dst[i];
        int pos = offs[d] + atomicAdd(&cur[d], 1);
        csrc[pos] = s;
        cw[pos] = dinv[s];
    }
}

// per-row insertion sort by src id -> CSR order deterministic across calls
__global__ void k_sortrows(int* __restrict__ csrc, float* __restrict__ cw,
                           const int* __restrict__ offs, const int* __restrict__ cnt, int n) {
    int i = blockIdx.x * blockDim.x + threadIdx.x;
    if (i >= n) return;
    int o = offs[i], m = cnt[i];
    for (int a = 1; a < m; ++a) {
        int s = csrc[o + a]; float w = cw[o + a];
        int b = a - 1;
        while (b >= 0 && csrc[o + b] > s) {
            csrc[o + b + 1] = csrc[o + b];
            cw[o + b + 1]   = cw[o + b];
            --b;
        }
        csrc[o + b + 1] = s;
        cw[o + b + 1]   = w;
    }
}

// ---------------- APPNP propagation: one wave32 per node, L2-resident gather ----------------
__global__ void k_prop(const float* __restrict__ zin, const float* __restrict__ h0,
                       const float* __restrict__ dinv, const int* __restrict__ offs,
                       const int* __restrict__ cnt, const int* __restrict__ csrc,
                       const float* __restrict__ cw, float* __restrict__ zout, int n) {
    int wid = (int)((blockIdx.x * blockDim.x + threadIdx.x) >> 5);
    if (wid >= n) return;
    int lane = threadIdx.x & 31;
    int fb = lane * 8;                        // 8 features/lane: two float4s
    int o = offs[wid], m = cnt[wid];
    float wd = dinv[wid];

    const f4* zs = (const f4*)(zin + (size_t)wid * HID + fb);
    f4 a0 = zs[0] * wd;                       // self-loop term (dinv[i]*z[i])
    f4 a1 = zs[1] * wd;
    for (int j = 0; j < m; ++j) {
        int s = csrc[o + j];
        float w = cw[o + j];
        const f4* p = (const f4*)(zin + (size_t)s * HID + fb);
        a0 += p[0] * w;
        a1 += p[1] * w;
    }
    float sc = (1.0f - ALPHA) * wd;
    const f4* hh = (const f4*)(h0 + (size_t)wid * HID + fb);
    f4* out = (f4*)(zout + (size_t)wid * HID + fb);
    out[0] = a0 * sc + hh[0] * ALPHA;
    out[1] = a1 * sc + hh[1] * ALPHA;
}

// ---------------- pooling over sorted batch (binary search bounds) ----------------
__global__ void k_bounds(const int* __restrict__ batch, int n, int g, int* __restrict__ lo) {
    int t = blockIdx.x * blockDim.x + threadIdx.x;
    if (t > g) return;
    int l = 0, h = n;
    while (l < h) { int m = (l + h) >> 1; if (batch[m] < t) l = m + 1; else h = m; }
    lo[t] = l;                                // lo[G] == n
}

__global__ void k_pool(const float* __restrict__ z, const int* __restrict__ lo,
                       float* __restrict__ p, int g) {
    int wid = (int)((blockIdx.x * blockDim.x + threadIdx.x) >> 5);
    if (wid >= g) return;
    int lane = threadIdx.x & 31;
    int fb = lane * 8;
    int a = lo[wid], b = lo[wid + 1];
    f4 s0 = {0.f, 0.f, 0.f, 0.f}, s1 = {0.f, 0.f, 0.f, 0.f};
    for (int r = a; r < b; ++r) {
        const f4* q = (const f4*)(z + (size_t)r * HID + fb);
        s0 += q[0]; s1 += q[1];
    }
    int c = b - a;
    float inv = 1.0f / (float)(c > 0 ? c : 1);
    f4* o = (f4*)(p + (size_t)wid * HID + fb);
    o[0] = s0 * inv;
    o[1] = s1 * inv;
}

// ---------------- tiny classifier head ----------------
__global__ void k_fc(const float* __restrict__ in, const float* __restrict__ W,
                     const float* __restrict__ b, float* __restrict__ out,
                     int M, int K, int Nout, int dorelu) {
    int total = M * Nout;
    for (int i = blockIdx.x * blockDim.x + threadIdx.x; i < total;
         i += gridDim.x * blockDim.x) {
        int row = i / Nout, o = i - row * Nout;
        const float* ir = in + (size_t)row * K;
        float s = b[o];
        for (int k = 0; k < K; ++k) s += ir[k] * W[(size_t)k * Nout + o];
        if (dorelu) s = fmaxf(s, 0.0f);
        out[i] = s;
    }
}

// ---------------- launch ----------------
extern "C" void kernel_launch(void* const* d_in, const int* in_sizes, int n_in,
                              void* d_out, int out_size, void* d_ws, size_t ws_size,
                              hipStream_t stream) {
    (void)n_in; (void)out_size; (void)ws_size;
    const float* x   = (const float*)d_in[0];
    const int*   ei  = (const int*)d_in[1];   // [2,E]: row0=src, row1=dst
    const int* batch = (const int*)d_in[2];
    const float* W1  = (const float*)d_in[3];
    const float* b1  = (const float*)d_in[4];
    const float* g1  = (const float*)d_in[5];
    const float* bt1 = (const float*)d_in[6];
    const float* W2  = (const float*)d_in[7];
    const float* b2  = (const float*)d_in[8];
    const float* g2  = (const float*)d_in[9];
    const float* bt2 = (const float*)d_in[10];
    const float* Wc1 = (const float*)d_in[11];
    const float* bc1 = (const float*)d_in[12];
    const float* Wc2 = (const float*)d_in[13];
    const float* bc2 = (const float*)d_in[14];
    const float* Wc3 = (const float*)d_in[15];
    const float* bc3 = (const float*)d_in[16];
    float* out = (float*)d_out;

    const int N = in_sizes[0] / DIM_IN;       // 100000 (multiple of 16)
    const int E = in_sizes[1] / 2;            // 1600000
    const int nb = (N + 255) / 256;           // stat/scan blocks

    // ---- workspace bump allocator (256B aligned slices) ----
    char* ws = (char*)d_ws;
    size_t off = 0;
    auto alloc = [&](size_t bytes) -> char* {
        char* p = ws + off;
        off += (bytes + 255) & ~(size_t)255;
        return p;
    };
    bf16*  xb    = (bf16*)alloc((size_t)N * KPAD1 * 2);      // later reused as z_a
    bf16*  w1t   = (bf16*)alloc((size_t)HID * KPAD1 * 2);
    bf16*  w2t   = (bf16*)alloc((size_t)HID * HID * 2);
    float* gbuf  = (float*)alloc((size_t)N * HID * 4);       // gemm out, later z_b
    bf16*  h1b   = (bf16*)alloc((size_t)N * HID * 2);
    float* h0    = (float*)alloc((size_t)N * HID * 4);
    int*   cntin = (int*)alloc((size_t)N * 4);
    float* dinv  = (float*)alloc((size_t)N * 4);
    int*   offs  = (int*)alloc((size_t)N * 4);
    int*   cur   = (int*)alloc((size_t)N * 4);
    int*   csrc  = (int*)alloc((size_t)E * 4);
    float* cw    = (float*)alloc((size_t)E * 4);
    int*   bsum  = (int*)alloc((size_t)nb * 4);
    float* ps    = (float*)alloc((size_t)nb * HID * 4);
    float* pq    = (float*)alloc((size_t)nb * HID * 4);
    float* mu    = (float*)alloc(HID * 4);
    float* rinv  = (float*)alloc(HID * 4);
    int*   glo   = (int*)alloc((N_GRAPH + 1) * 4);
    float* pool  = (float*)alloc((size_t)N_GRAPH * HID * 4);
    float* c1    = (float*)alloc((size_t)N_GRAPH * HID * 4);
    float* c2    = (float*)alloc((size_t)N_GRAPH * (HID / 2) * 4);
    float* za    = (float*)xb;      // 160MB slice >= 102.4MB needed
    float* zb    = gbuf;

    // ---- encoder: convert/pack, 2x WMMA GEMM + BN + ReLU ----
    k_cvt_pad_x<<<4096, 256, 0, stream>>>(x, xb, N * KPAD1);
    k_packT<<<(HID * KPAD1 + 255) / 256, 256, 0, stream>>>(W1, w1t, DIM_IN, HID, KPAD1);
    k_packT<<<(HID * HID + 255) / 256, 256, 0, stream>>>(W2, w2t, HID, HID, HID);

    int waves = (N / 16) * 4;                       // 16-row blocks x 4 n-groups
    int gemmBlocks = (waves * 32 + 255) / 256;
    k_gemm_wmma<<<gemmBlocks, 256, 0, stream>>>(xb, KPAD1, w1t, KPAD1, b1, gbuf, N, KPAD1 / 32);
    k_colstats_part<<<nb, 256, 0, stream>>>(gbuf, N, ps, pq);
    k_colstats_final<<<1, 256, 0, stream>>>(ps, pq, nb, N, mu, rinv);
    k_bnrelu_bf16<<<4096, 256, 0, stream>>>(gbuf, mu, rinv, g1, bt1, h1b, N * HID);

    k_gemm_wmma<<<gemmBlocks, 256, 0, stream>>>(h1b, HID, w2t, HID, b2, gbuf, N, HID / 32);
    k_colstats_part<<<nb, 256, 0, stream>>>(gbuf, N, ps, pq);
    k_colstats_final<<<1, 256, 0, stream>>>(ps, pq, nb, N, mu, rinv);
    k_bnrelu_f32<<<4096, 256, 0, stream>>>(gbuf, mu, rinv, g2, bt2, h0, N * HID);

    // ---- gcn_norm + CSR (dst-indexed), deterministic ----
    k_zero_i32<<<(N + 255) / 256, 256, 0, stream>>>(cntin, N);
    k_zero_i32<<<(N + 255) / 256, 256, 0, stream>>>(cur, N);
    k_count_in<<<2048, 256, 0, stream>>>(ei + E, E, cntin);
    k_dinv<<<(N + 255) / 256, 256, 0, stream>>>(cntin, dinv, N);
    k_scanA<<<nb, 256, 0, stream>>>(cntin, N, offs, bsum);
    k_scanB<<<1, 1, 0, stream>>>(bsum, nb);
    k_scanC<<<(N + 255) / 256, 256, 0, stream>>>(offs, bsum, N);
    k_fill<<<2048, 256, 0, stream>>>(ei, ei + E, E, offs, cur, dinv, csrc, cw);
    k_sortrows<<<(N + 255) / 256, 256, 0, stream>>>(csrc, cw, offs, cntin, N);

    // ---- K=10 propagation steps (ping-pong, wave per node) ----
    const float* zi = h0;
    float* zo = zb;
    int pblocks = (N * 32 + 255) / 256;
    for (int it = 0; it < KSTEPS; ++it) {
        k_prop<<<pblocks, 256, 0, stream>>>(zi, h0, dinv, offs, cntin, csrc, cw, zo, N);
        zi = zo;
        zo = (zo == zb) ? za : zb;
    }

    // ---- global mean pool + classifier head ----
    k_bounds<<<(N_GRAPH + 1 + 255) / 256, 256, 0, stream>>>(batch, N, N_GRAPH, glo);
    k_pool<<<(N_GRAPH * 32 + 255) / 256, 256, 0, stream>>>(zi, glo, pool, N_GRAPH);
    k_fc<<<(N_GRAPH * HID + 255) / 256, 256, 0, stream>>>(pool, Wc1, bc1, c1, N_GRAPH, HID, HID, 1);
    k_fc<<<(N_GRAPH * (HID / 2) + 255) / 256, 256, 0, stream>>>(c1, Wc2, bc2, c2, N_GRAPH, HID, HID / 2, 1);
    k_fc<<<(N_GRAPH + 255) / 256, 256, 0, stream>>>(c2, Wc3, bc3, out, N_GRAPH, HID / 2, 1, 0);
}
